// DNDF_71270687310068
// MI455X (gfx1250) — compile-verified, matrix operand order checked by reading, not run on hardware
//
#include <hip/hip_runtime.h>
#include <hip/hip_bf16.h>

// ---------------------------------------------------------------------------
// DNDF forward, fused for gfx1250 (MI455X): bf16 WMMA for the big GEMM
// (x @ W^T), sigmoid + soft-route products, then a second WMMA chain
// (routes @ softmax(leafs)/64) accumulated in f32.
// ---------------------------------------------------------------------------

#define BATCH    8192
#define IN_DIM   512
#define NUM_TREE 64
#define NUM_LEAF 32          // 2^(depth+1), depth=4
#define NUM_INT  31          // internals per tree (padded to 32)
#define OUT_DIM  16
#define XL_STRIDE 520        // bf16 elems per LDS x-row (16B aligned, bank-skewed)

typedef __attribute__((ext_vector_type(8)))  float  v8f;
typedef __attribute__((ext_vector_type(16))) __bf16 v16bf;

union Frag16 {
    v16bf v;
    unsigned short u[16];
    uint4 q[2];
};

__device__ __forceinline__ unsigned short f2bf(float f) {
    unsigned int u = __float_as_uint(f);
    unsigned int r = u + 0x7FFFu + ((u >> 16) & 1u);   // round-to-nearest-even
    return (unsigned short)(r >> 16);
}

__device__ __forceinline__ float sigf(float x) {
    return 1.0f / (1.0f + __expf(-x));
}

// ---------------------------------------------------------------------------
// Pack W [64*31, 512] f32 -> bf16 B-fragments.
// Layout: Wpf[t][kk][ct][lane][e], t<64, kk<16 (K step of 32), ct<2 (col tile),
// lane<32, e<16.  Lane holds matrix column n = ct*16 + (lane&15);
// elements 0..7 = K {koff..koff+7}, 8..15 = K {16+koff..}, koff=(lane>>4)*8.
// Rows n==31 of each tree are zero padding.
// ---------------------------------------------------------------------------
__global__ void dndf_pack_w(const float* __restrict__ W,
                            unsigned short* __restrict__ Wpf) {
    int idx = blockIdx.x * 256 + threadIdx.x;
    const int total = NUM_TREE * 16 * 2 * 32 * 16;   // 1,048,576
    if (idx >= total) return;
    int e    = idx & 15;
    int lane = (idx >> 4) & 31;
    int ct   = (idx >> 9) & 1;
    int kk   = (idx >> 10) & 15;
    int t    = idx >> 14;
    int n    = ct * 16 + (lane & 15);
    int koff = (lane >> 4) * 8;
    int k    = kk * 32 + ((e < 8) ? (koff + e) : (16 + koff + (e - 8)));
    unsigned short v = 0;
    if (n < NUM_INT) v = f2bf(W[(t * NUM_INT + n) * IN_DIM + k]);
    Wpf[idx] = v;
}

// ---------------------------------------------------------------------------
// Pack softmax(leafs)/64 -> bf16 B-fragments smf[t][lane][16]
// (B matrix is [32 leaves x 16 outs]); also padded bias bp[2048].
// One thread per (t, lane): 2048 threads.
// ---------------------------------------------------------------------------
__global__ void dndf_pack_sm(const float* __restrict__ leafs,
                             const float* __restrict__ b,
                             unsigned short* __restrict__ smf,
                             float* __restrict__ bp) {
    int idx = blockIdx.x * 256 + threadIdx.x;
    if (idx >= NUM_TREE * 32) return;
    int lane = idx & 31;
    int t    = idx >> 5;
    int n    = lane & 15;          // output column
    int koff = (lane >> 4) * 8;
    for (int e = 0; e < 16; ++e) {
        int leaf = (e < 8) ? (koff + e) : (16 + koff + (e - 8));
        const float* row = leafs + (t * NUM_LEAF + leaf) * OUT_DIM;
        float mx = row[0];
        #pragma unroll
        for (int o = 1; o < OUT_DIM; ++o) mx = fmaxf(mx, row[o]);
        float sum = 0.f;
        #pragma unroll
        for (int o = 0; o < OUT_DIM; ++o) sum += __expf(row[o] - mx);
        float val = __expf(row[n] - mx) / (sum * (float)NUM_TREE);
        smf[idx * 16 + e] = f2bf(val);
    }
    int i = idx & 31;   // padded bias: bp[t*32 + i]
    bp[idx] = (i < NUM_INT) ? b[(idx >> 5) * NUM_INT + i] : 0.0f;
}

// ---------------------------------------------------------------------------
// Main fused kernel: one block = 16 batch rows, 8 waves; wave w owns trees
// 8w..8w+7 end-to-end; cross-wave reduce in LDS; single deterministic store.
// ---------------------------------------------------------------------------
__global__ __launch_bounds__(256) void dndf_main(
        const float* __restrict__ x,
        const unsigned short* __restrict__ Wpf,
        const unsigned short* __restrict__ smf,
        const float* __restrict__ bp,
        float* __restrict__ out) {

    __shared__ unsigned short xtile[16 * XL_STRIDE];   // 16,640 B bf16 x tile
    __shared__ float pbuf[8][16][33];                   // per-wave sigmoid probs
    __shared__ float red[8][16][16];                    // cross-wave out reduce

    const int tid  = threadIdx.x;
    const int wave = tid >> 5;
    const int lane = tid & 31;
    const int row0 = blockIdx.x * 16;

    // --- stage x tile (16 x 512 f32 -> bf16) into LDS ---
    for (int i = tid; i < 16 * (IN_DIM / 4); i += 256) {
        int r  = i / (IN_DIM / 4);
        int c4 = i % (IN_DIM / 4);
        float4 f = ((const float4*)(x + (size_t)(row0 + r) * IN_DIM))[c4];
        int o = r * XL_STRIDE + c4 * 4;
        xtile[o + 0] = f2bf(f.x);
        xtile[o + 1] = f2bf(f.y);
        xtile[o + 2] = f2bf(f.z);
        xtile[o + 3] = f2bf(f.w);
    }
    __syncthreads();

    const int ln    = lane & 15;          // A-frag row / B-frag column
    const int khalf = (lane >> 4) * 8;    // K sub-chunk select
    const int mrow  = (lane >> 4) * 8;    // C-frag row base

    v8f out_acc = {};

    for (int tl = 0; tl < 8; ++tl) {
        const int t = wave * 8 + tl;
        v8f c0 = {};
        v8f c1 = {};
        const unsigned short* wp = Wpf + (size_t)t * (16 * 2 * 32 * 16);

        for (int kk = 0; kk < 16; ++kk) {
            // A fragment: x tile rows, from LDS (two ds_load_b128 per lane)
            Frag16 a;
            const unsigned short* xr = &xtile[ln * XL_STRIDE + kk * 32 + khalf];
            a.q[0] = *(const uint4*)(xr);
            a.q[1] = *(const uint4*)(xr + 16);
            // B fragments: pre-swizzled W, contiguous 32B per lane
            Frag16 b0, b1;
            const uint4* wb0 = (const uint4*)(wp + (((size_t)kk * 2 + 0) * 32 + lane) * 16);
            const uint4* wb1 = (const uint4*)(wp + (((size_t)kk * 2 + 1) * 32 + lane) * 16);
            b0.q[0] = wb0[0]; b0.q[1] = wb0[1];
            b1.q[0] = wb1[0]; b1.q[1] = wb1[1];

            c0 = __builtin_amdgcn_wmma_f32_16x16x32_bf16(
                    false, a.v, false, b0.v, (short)0, c0, false, false);
            c1 = __builtin_amdgcn_wmma_f32_16x16x32_bf16(
                    false, a.v, false, b1.v, (short)0, c1, false, false);
        }

        // bias + sigmoid, stage probabilities for cross-lane route products
        const float bias0 = bp[t * 32 + ln];
        const float bias1 = bp[t * 32 + 16 + ln];
        #pragma unroll
        for (int v = 0; v < 8; ++v) {
            pbuf[wave][mrow + v][ln]      = sigf(c0[v] + bias0);
            pbuf[wave][mrow + v][16 + ln] = sigf(c1[v] + bias1);
        }
        __syncthreads();

        // route products -> bf16 A fragment (rows = batch, K = 32 leaves)
        Frag16 r;
        #pragma unroll
        for (int e = 0; e < 16; ++e) {
            int leaf = (e < 8) ? (khalf + e) : (16 + khalf + (e - 8));
            float prod = 1.f;
            #pragma unroll
            for (int d = 0; d <= 4; ++d) {
                int node = ((1 << d) - 1) + (leaf >> (5 - d));
                float p  = pbuf[wave][ln][node];
                prod *= (((leaf >> (4 - d)) & 1) ? (1.f - p) : p);
            }
            r.u[e] = f2bf(prod);
        }

        // leaf-softmax B fragment and output WMMA chain
        Frag16 s;
        const uint4* sp = (const uint4*)(smf + ((size_t)t * 32 + lane) * 16);
        s.q[0] = sp[0]; s.q[1] = sp[1];
        out_acc = __builtin_amdgcn_wmma_f32_16x16x32_bf16(
                    false, r.v, false, s.v, (short)0, out_acc, false, false);
        __syncthreads();   // protect pbuf against next tree's writes
    }

    // cross-wave reduction of the 16x16 output tile
    #pragma unroll
    for (int v = 0; v < 8; ++v) red[wave][mrow + v][ln] = out_acc[v];
    __syncthreads();

    const int rrow = tid >> 4;
    const int col  = tid & 15;
    float sum = 0.f;
    #pragma unroll
    for (int w = 0; w < 8; ++w) sum += red[w][rrow][col];
    out[(size_t)(row0 + rrow) * OUT_DIM + col] = sum;
}

// ---------------------------------------------------------------------------
extern "C" void kernel_launch(void* const* d_in, const int* in_sizes, int n_in,
                              void* d_out, int out_size, void* d_ws, size_t ws_size,
                              hipStream_t stream) {
    const float* x     = (const float*)d_in[0];
    const float* W     = (const float*)d_in[1];
    const float* b     = (const float*)d_in[2];
    const float* leafs = (const float*)d_in[3];
    float* out = (float*)d_out;

    // workspace layout: [Wpf bf16 2MB][smf bf16 64KB][bp f32 8KB]
    unsigned short* Wpf = (unsigned short*)d_ws;
    unsigned short* smf = (unsigned short*)((char*)d_ws + (size_t)2 * 1024 * 1024);
    float*          bpp = (float*)((char*)d_ws + (size_t)2 * 1024 * 1024 + 64 * 1024);

    dndf_pack_w <<<4096, 256, 0, stream>>>(W, Wpf);
    dndf_pack_sm<<<8,    256, 0, stream>>>(leafs, b, smf, bpp);
    dndf_main   <<<BATCH / 16, 256, 0, stream>>>(x, Wpf, smf, bpp, out);
}
